// MultiHeadAttention_46428596470221
// MI455X (gfx1250) — matrix operand order repaired and checked by reading, not compiled, not run on hardware
//
#include <hip/hip_runtime.h>

#define B_  2
#define T_  2048
#define D_  1024
#define H_  16
#define DH_ 64

typedef __bf16 bf16_t;
typedef __attribute__((ext_vector_type(16))) __bf16 v16bf;
typedef __attribute__((ext_vector_type(8)))  __bf16 v8bf;
typedef __attribute__((ext_vector_type(8)))  float  v8f;
typedef __attribute__((ext_vector_type(4)))  unsigned int v4u;
typedef __attribute__((ext_vector_type(8)))  int v8i;
typedef __attribute__((ext_vector_type(4)))  int v4i;

#if __has_include(<hip/amd_detail/amd_gfx1250_TDM.h>)
#define TDM_6ARG 1
#endif

union V16U { v16bf v; v8bf h[2]; };

// round-to-nearest-even f32 -> bf16
static __device__ __forceinline__ bf16_t f2bf(float f) {
    unsigned u = __float_as_uint(f);
    u = (u + 0x7FFFu + ((u >> 16) & 1u)) >> 16;
    unsigned short s = (unsigned short)u;
    bf16_t o;
    __builtin_memcpy(&o, &s, 2);
    return o;
}

static __device__ __forceinline__ v8f wmma_bf16(v16bf a, v16bf b, v8f c) {
    return __builtin_amdgcn_wmma_f32_16x16x32_bf16(false, a, false, b, (short)0, c,
                                                   false, false);
}

// A-matrix (16x32, MxK) loader. Row-major source, row = lane&15.
// element e -> K = kbase + e + 8*((e>>3) + (lane>>4))  => two 16B chunks.
static __device__ __forceinline__ v16bf load_A_tile(const bf16_t* base, int ld,
                                                    int kbase, int lane) {
    const int m = lane & 15;
    const int g = lane >> 4;
    const bf16_t* p = base + (size_t)m * ld + kbase + 8 * g;
    V16U u;
    u.h[0] = *(const v8bf*)(p);       // K = kbase + 8g .. +7
    u.h[1] = *(const v8bf*)(p + 16);  // K = kbase + 16 + 8g .. +7
    return u.v;
}

// B-matrix (32x16, KxN) loader. Source rows indexed by N (ld = row stride),
// contiguous along K. lane: N = lane&15; element e -> K = kbase + e + 8*((e>>3) + (lane>>4)).
static __device__ __forceinline__ v16bf load_B_tile(const bf16_t* base, int ld,
                                                    int kbase, int lane) {
    const int n = lane & 15;
    const int g = lane >> 4;
    const bf16_t* p = base + (size_t)n * ld + kbase + 8 * g;
    V16U u;
    u.h[0] = *(const v8bf*)(p);       // K = kbase + 8g .. +7
    u.h[1] = *(const v8bf*)(p + 16);  // K = kbase + 16 + 8g .. +7
    return u.v;
}

// ---------------------------------------------------------------- TDM issue
// 2D tile DMA global->LDS. D# packing per CDNA5 ISA.
// tile: tile1 rows x tile0 contiguous elements, source row stride = stride0
// (all counts in 2-byte elements; data_size field = 1 -> 2 bytes).
static __device__ __forceinline__ void tdm_load_2d(unsigned lds_off,
                                                   const void* gptr,
                                                   int tile0, int tile1,
                                                   int stride0) {
    const unsigned long long ga = (unsigned long long)gptr;
    const unsigned dimBig = 1u << 20;     // generous tensor_dim for OOB checks
    v4u g0;
    g0[0] = 1u;                                               // count=1, user mode
    g0[1] = lds_off;                                          // lds_addr (bytes)
    g0[2] = (unsigned)(ga & 0xFFFFFFFFull);                   // global_addr[31:0]
    g0[3] = (unsigned)((ga >> 32) & 0x01FFFFFFull)            // global_addr[56:32]
          | 0x80000000u;                                      // type=2 ("image")
    v8i g1;
    g1[0] = 0x00010000;                                       // data_size=1 (2B)
    g1[1] = (int)((dimBig & 0xFFFFu) << 16);                  // tensor_dim0[15:0]
    g1[2] = (int)(((dimBig >> 16) & 0xFFFFu)                  // tensor_dim0[31:16]
          | ((dimBig & 0xFFFFu) << 16));                      // tensor_dim1[15:0]
    g1[3] = (int)(((dimBig >> 16) & 0xFFFFu)                  // tensor_dim1[31:16]
          | ((unsigned)tile0 << 16));                         // tile_dim0
    g1[4] = tile1 & 0xFFFF;                                   // tile_dim1 (tile2=0)
    g1[5] = stride0;                                          // tensor_dim0_stride
    g1[6] = 0;
    g1[7] = 0;
    const v4i z4 = {0, 0, 0, 0};
#ifdef TDM_6ARG
    const v8i z8 = {0, 0, 0, 0, 0, 0, 0, 0};
    __builtin_amdgcn_tensor_load_to_lds(g0, g1, z4, z4, z8, 0);
#else
    __builtin_amdgcn_tensor_load_to_lds(g0, g1, z4, z4, 0);
#endif
}

// ---------------------------------------------------------------- cast stage
__global__ __launch_bounds__(256) void cast_f32_to_bf16(const float* __restrict__ src,
                                                        bf16_t* __restrict__ dst,
                                                        int n4) {
    int i = blockIdx.x * 256 + threadIdx.x;
    if (i < n4) {
        float4 f = ((const float4*)src)[i];
        bf16_t* d = dst + 4 * (size_t)i;
        d[0] = f2bf(f.x); d[1] = f2bf(f.y); d[2] = f2bf(f.z); d[3] = f2bf(f.w);
    }
}

// ------------------------------------------------- TDM-staged pipelined GEMM
// y[row, e] = sum_d a[row, d] * W[e, d].  Block = 4 waves = 128x64 output tile,
// each wave 32x64 (8 accumulators).  Wave0 DMAs the 128x32 A panel, wave1 the
// 64x32 B panel into double-buffered LDS; one barrier per K-step, DMA for step
// k+1 overlaps the 8 WMMAs of step k.
// MODE 0: bf16 dst[b,h,t,dh]   MODE 1: bf16 dst[b,h,dh,t]   MODE 2: f32 row-major
template <int MODE>
__global__ __launch_bounds__(128) void gemm_tdm(const bf16_t* __restrict__ a,
                                                const bf16_t* __restrict__ w,
                                                void* __restrict__ dstv) {
    __shared__ bf16_t ldsA[2][128 * 32];   // 16 KiB
    __shared__ bf16_t ldsB[2][64 * 32];    //  8 KiB

    const int lane = threadIdx.x & 31;
    const int wave = threadIdx.x >> 5;
    const int rowBlk = blockIdx.x * 128;
    const int col0 = blockIdx.y * 64;

    const bf16_t* aBlk = a + (size_t)rowBlk * D_;
    const bf16_t* wBlk = w + (size_t)col0 * D_;

    v8f acc[2][4] = {};

    if (wave == 0)
        tdm_load_2d((unsigned)(unsigned long long)&ldsA[0][0], aBlk, 32, 128, D_);
    if (wave == 1)
        tdm_load_2d((unsigned)(unsigned long long)&ldsB[0][0], wBlk, 32, 64, D_);

    int cur = 0;
    for (int ki = 0; ki < D_ / 32; ++ki) {
        __builtin_amdgcn_s_wait_tensorcnt(0);   // producers: DMA into buf[cur] done
        __syncthreads();                        // publish to all 4 waves
        if (ki + 1 < D_ / 32) {                 // prefetch next panel (other buffer)
            const int kn = (ki + 1) * 32;
            if (wave == 0)
                tdm_load_2d((unsigned)(unsigned long long)&ldsA[cur ^ 1][0],
                            aBlk + kn, 32, 128, D_);
            if (wave == 1)
                tdm_load_2d((unsigned)(unsigned long long)&ldsB[cur ^ 1][0],
                            wBlk + kn, 32, 64, D_);
        }
        const v16bf a0 = load_A_tile(&ldsA[cur][(wave * 32 + 0) * 32], 32, 0, lane);
        const v16bf a1 = load_A_tile(&ldsA[cur][(wave * 32 + 16) * 32], 32, 0, lane);
#pragma unroll
        for (int n = 0; n < 4; ++n) {
            const v16bf b = load_B_tile(&ldsB[cur][(n * 16) * 32], 32, 0, lane);
            acc[0][n] = wmma_bf16(a0, b, acc[0][n]);
            acc[1][n] = wmma_bf16(a1, b, acc[1][n]);
        }
        cur ^= 1;
    }

    const int g = lane >> 4, nl = lane & 15;
#pragma unroll
    for (int m = 0; m < 2; ++m)
#pragma unroll
        for (int n = 0; n < 4; ++n) {
            const int e = col0 + n * 16 + nl;
#pragma unroll
            for (int r = 0; r < 8; ++r) {
                const int row = rowBlk + wave * 32 + m * 16 + r + 8 * g;
                if (MODE == 2) {
                    ((float*)dstv)[(size_t)row * D_ + e] = acc[m][n][r];
                } else {
                    const int b = row >> 11;           // row / T_
                    const int t = row & (T_ - 1);
                    const int h = e >> 6, dh = e & (DH_ - 1);
                    const bf16_t val = f2bf(acc[m][n][r]);
                    if (MODE == 0)
                        ((bf16_t*)dstv)[(((size_t)(b * H_ + h)) * T_ + t) * DH_ + dh] = val;
                    else
                        ((bf16_t*)dstv)[(((size_t)(b * H_ + h)) * DH_ + dh) * T_ + t] = val;
                }
            }
        }
}

// ------------------------------------------------------ flash attention core
// One wave per (b,h, 16-query tile). K,Q in [B,H,T,DH] bf16; V in [B,H,DH,T].
__global__ __launch_bounds__(32) void attn_kernel(const bf16_t* __restrict__ q,
                                                  const bf16_t* __restrict__ k,
                                                  const bf16_t* __restrict__ vt,
                                                  bf16_t* __restrict__ ctx) {
    __shared__ bf16_t pbuf[16 * 32];          // P bounce buffer (D-layout -> A-layout)

    const int lane = threadIdx.x & 31;
    const int bh = blockIdx.x;                // b*H + h
    const int q0 = blockIdx.y * 16;
    const int b = bh >> 4, h = bh & 15;
    const int g = lane >> 4, nl = lane & 15;

    const bf16_t* qBase = q  + ((size_t)bh * T_ + q0) * DH_;
    const bf16_t* kBase = k  +  (size_t)bh * T_ * DH_;
    const bf16_t* vBase = vt +  (size_t)bh * DH_ * T_;

    const v16bf aQ0 = load_A_tile(qBase, DH_, 0,  lane);
    const v16bf aQ1 = load_A_tile(qBase, DH_, 32, lane);

    v8f O[4] = {};
    float m_r[8], l_r[8];
#pragma unroll
    for (int r = 0; r < 8; ++r) { m_r[r] = -__builtin_inff(); l_r[r] = 0.0f; }

    for (int k0 = 0; k0 < q0 + 16; k0 += 32) {   // causal: only blocks with k0 <= q0
        v8f s[2] = {};
#pragma unroll
        for (int n = 0; n < 2; ++n) {
            const bf16_t* kb = kBase + (size_t)(k0 + n * 16) * DH_;
            v16bf b0 = load_B_tile(kb, DH_, 0,  lane);
            v16bf b1 = load_B_tile(kb, DH_, 32, lane);
            s[n] = wmma_bf16(aQ0, b0, s[n]);
            s[n] = wmma_bf16(aQ1, b1, s[n]);
        }
#pragma unroll
        for (int n = 0; n < 2; ++n)
#pragma unroll
            for (int r = 0; r < 8; ++r) {
                const int qrow = q0 + r + 8 * g;
                const int key  = k0 + n * 16 + nl;
                const float val = s[n][r] * 0.125f;   // 1/sqrt(64)
                s[n][r] = (key <= qrow) ? val : -__builtin_inff();
            }
#pragma unroll
        for (int r = 0; r < 8; ++r) {
            float vmax = fmaxf(s[0][r], s[1][r]);
#pragma unroll
            for (int mm = 1; mm <= 8; mm <<= 1)
                vmax = fmaxf(vmax, __shfl_xor(vmax, mm, 32));
            const float m_new = fmaxf(m_r[r], vmax);
            const float corr  = __expf(m_r[r] - m_new);
            float psum = 0.0f;
#pragma unroll
            for (int n = 0; n < 2; ++n) {
                const float p = __expf(s[n][r] - m_new);
                s[n][r] = p;
                psum += p;
            }
#pragma unroll
            for (int mm = 1; mm <= 8; mm <<= 1)
                psum += __shfl_xor(psum, mm, 32);
            l_r[r] = l_r[r] * corr + psum;
            m_r[r] = m_new;
#pragma unroll
            for (int n = 0; n < 4; ++n) O[n][r] *= corr;
        }
#pragma unroll
        for (int n = 0; n < 2; ++n)
#pragma unroll
            for (int r = 0; r < 8; ++r)
                pbuf[(r + 8 * g) * 32 + n * 16 + nl] = f2bf(s[n][r]);
        asm volatile("s_wait_dscnt 0" ::: "memory");   // intra-wave DS store->load
        const v16bf aP = load_A_tile(pbuf, 32, 0, lane);
#pragma unroll
        for (int n = 0; n < 4; ++n) {
            v16bf bV = load_B_tile(vBase + (size_t)(n * 16) * T_, T_, k0, lane);
            O[n] = wmma_bf16(aP, bV, O[n]);
        }
    }

#pragma unroll
    for (int r = 0; r < 8; ++r) {
        const float inv = 1.0f / l_r[r];
        const int t = q0 + r + 8 * g;
        const size_t base = ((size_t)b * T_ + t) * D_ + h * DH_;
#pragma unroll
        for (int n = 0; n < 4; ++n)
            ctx[base + n * 16 + nl] = f2bf(O[n][r] * inv);
    }
}

// ---------------------------------------------------------------- launcher
extern "C" void kernel_launch(void* const* d_in, const int* in_sizes, int n_in,
                              void* d_out, int out_size, void* d_ws, size_t ws_size,
                              hipStream_t stream) {
    (void)in_sizes; (void)n_in; (void)out_size; (void)ws_size;
    const float* x  = (const float*)d_in[0];
    const float* wq = (const float*)d_in[1];
    const float* wk = (const float*)d_in[2];
    const float* wv = (const float*)d_in[3];
    const float* wo = (const float*)d_in[4];
    float* out = (float*)d_out;

    const size_t nX = (size_t)B_ * T_ * D_;   // 4 Mi elems
    const size_t nW = (size_t)D_ * D_;        // 1 Mi elems

    bf16_t* ws   = (bf16_t*)d_ws;
    bf16_t* xbf  = ws; ws += nX;
    bf16_t* wqb  = ws; ws += nW;
    bf16_t* wkb  = ws; ws += nW;
    bf16_t* wvb  = ws; ws += nW;
    bf16_t* wob  = ws; ws += nW;
    bf16_t* qb   = ws; ws += nX;
    bf16_t* kb   = ws; ws += nX;
    bf16_t* vtb  = ws; ws += nX;
    bf16_t* ctxb = ws; ws += nX;

    cast_f32_to_bf16<<<(int)(nX / 4 / 256), 256, 0, stream>>>(x,  xbf, (int)(nX / 4));
    cast_f32_to_bf16<<<(int)(nW / 4 / 256), 256, 0, stream>>>(wq, wqb, (int)(nW / 4));
    cast_f32_to_bf16<<<(int)(nW / 4 / 256), 256, 0, stream>>>(wk, wkb, (int)(nW / 4));
    cast_f32_to_bf16<<<(int)(nW / 4 / 256), 256, 0, stream>>>(wv, wvb, (int)(nW / 4));
    cast_f32_to_bf16<<<(int)(nW / 4 / 256), 256, 0, stream>>>(wo, wob, (int)(nW / 4));

    dim3 gGemm(B_ * T_ / 128, D_ / 64);   // (32, 16), 128 threads = 4 waves
    gemm_tdm<0><<<gGemm, 128, 0, stream>>>(xbf, wqb, qb);
    gemm_tdm<0><<<gGemm, 128, 0, stream>>>(xbf, wkb, kb);
    gemm_tdm<1><<<gGemm, 128, 0, stream>>>(xbf, wvb, vtb);

    attn_kernel<<<dim3(B_ * H_, T_ / 16), 32, 0, stream>>>(qb, kb, vtb, ctxb);

    gemm_tdm<2><<<gGemm, 128, 0, stream>>>(ctxb, wob, out);
}
